// GraphTransformer_71829033058899
// MI455X (gfx1250) — matrix operand order repaired and checked by reading, not compiled, not run on hardware
//
#include <hip/hip_runtime.h>
#include <hip/hip_bf16.h>
#include <math.h>

// ---------------------------------------------------------------------------
// GraphTransformer on MI455X (gfx1250): all GEMMs via v_wmma_f32_16x16x32_bf16
// LDS double-buffered 128x64 block tiles, 32x64 per wave (8 WMMA / K-step).
// ---------------------------------------------------------------------------

#define N_NODES 4096
#define DIN     512
#define DH      512
#define DOUT    256
#define NLAYERS 4

// LDS row stride (elements) for 32-K slices: 40 elems = 80 B = 20 banks ->
// 16 fragment lanes hit 16 distinct banks; keeps 16B alignment for b128 ops.
#define LDS_STRIDE 40

typedef __bf16 bf16_t;
typedef __attribute__((ext_vector_type(16))) __bf16 bfx16;
typedef __attribute__((ext_vector_type(8)))  __bf16 bfx8;
typedef __attribute__((ext_vector_type(8)))  float  fx8;

// ---------------- conversion / transpose helpers ----------------

__global__ void k_f32_to_bf16(const float* __restrict__ s, bf16_t* __restrict__ d, int n) {
    int i = blockIdx.x * blockDim.x + threadIdx.x;
    if (i < n) d[i] = (bf16_t)s[i];
}

// src: batch of `count` row-major [K x N] f32 matrices -> dst: [N x K] bf16 (transposed)
__global__ void k_transpose_bf16(const float* __restrict__ src, bf16_t* __restrict__ dst,
                                 int K, int N, int count) {
    int i = blockIdx.x * blockDim.x + threadIdx.x;
    int per = K * N;
    if (i < per * count) {
        int b = i / per, r = i % per;
        int k = r / N, n = r % N;
        dst[(size_t)b * per + (size_t)n * K + k] = (bf16_t)src[i];
    }
}

// ---------------- WMMA nt-GEMM: C[M x Nt] = A[M x K] * Bt[Nt x K]^T ----------------
// A, Bt bf16 row-major (lda = ldb = K). Block: 4 waves, tile 128(M) x 64(N).
// Wave: 32 x 64 -> 8 f32 accumulators. A/B K-slices double-buffered in LDS.
__global__ void __launch_bounds__(128)
k_gemm_nt(const bf16_t* __restrict__ A, const bf16_t* __restrict__ Bt,
          const float* __restrict__ bias,
          float* __restrict__ Cf, bf16_t* __restrict__ Cb, bf16_t* __restrict__ CbT,
          int M, int Nt, int K, float scale, int relu, int addpe) {
    __shared__ bf16_t ldsA[2][128 * LDS_STRIDE];   // 128 rows x 32 K (padded)
    __shared__ bf16_t ldsB[2][ 64 * LDS_STRIDE];   //  64 cols x 32 K (padded)

    const int tid  = threadIdx.x;
    const int lane = tid & 31;
    const int wave = tid >> 5;
    const int r    = lane & 15;   // fragment row (A) / output column (B)
    const int hi   = lane >> 4;   // K-half select per CDNA5 16-bit fragment layout

    const int tileM = blockIdx.x * 128;
    const int tileN = blockIdx.y * 64;

    fx8 acc[2][4] = {};

    // ---- cooperative staging: one 16B chunk = 8 bf16 of one row's K-slice ----
    auto stage = [&](int buf, int k0) {
        #pragma unroll
        for (int i = 0; i < 4; ++i) {              // A: 128 rows x 4 segs
            int e = tid + i * 128;
            int row = e >> 2, seg = e & 3;
            *(bfx8*)&ldsA[buf][row * LDS_STRIDE + seg * 8] =
                *(const bfx8*)(A + (size_t)(tileM + row) * K + k0 + seg * 8);
        }
        #pragma unroll
        for (int i = 0; i < 2; ++i) {              // B: 64 rows x 4 segs
            int e = tid + i * 128;
            int row = e >> 2, seg = e & 3;
            *(bfx8*)&ldsB[buf][row * LDS_STRIDE + seg * 8] =
                *(const bfx8*)(Bt + (size_t)(tileN + row) * K + k0 + seg * 8);
        }
    };

    stage(0, 0);
    __syncthreads();

    for (int k0 = 0; k0 < K; k0 += 32) {
        const int cur = (k0 >> 5) & 1;
        if (k0 + 32 < K) stage(cur ^ 1, k0 + 32);

        // A fragments: rows wave*32 + sub*16 + r; K {hi*8..+7, 16+hi*8..+7}
        #pragma unroll
        for (int sub = 0; sub < 2; ++sub) {
            const bf16_t* ar = &ldsA[cur][(wave * 32 + sub * 16 + r) * LDS_STRIDE];
            bfx8 alo = *(const bfx8*)(ar + hi * 8);
            bfx8 ahi = *(const bfx8*)(ar + 16 + hi * 8);
            bfx16 a = __builtin_shufflevector(alo, ahi,
                        0,1,2,3,4,5,6,7,8,9,10,11,12,13,14,15);
            #pragma unroll
            for (int j = 0; j < 4; ++j) {
                const bf16_t* br = &ldsB[cur][(j * 16 + r) * LDS_STRIDE];
                bfx8 blo = *(const bfx8*)(br + hi * 8);
                bfx8 bhi = *(const bfx8*)(br + 16 + hi * 8);
                bfx16 b = __builtin_shufflevector(blo, bhi,
                            0,1,2,3,4,5,6,7,8,9,10,11,12,13,14,15);
                acc[sub][j] = __builtin_amdgcn_wmma_f32_16x16x32_bf16(
                    false, a, false, b, (short)0, acc[sub][j], false, false);
            }
        }
        __syncthreads();
    }

    // C layout: VGPR e, lane L -> (M = e + 8*(L>>4), N = L&15)
    #pragma unroll
    for (int sub = 0; sub < 2; ++sub) {
        #pragma unroll
        for (int j = 0; j < 4; ++j) {
            const int n = tileN + j * 16 + r;
            float bv = bias ? bias[n] : 0.0f;
            if (addpe && (n & 1)) bv += 1.0f;   // pe[0]: sin(0)=0 (even), cos(0)=1 (odd)
            #pragma unroll
            for (int e = 0; e < 8; ++e) {
                const int m = tileM + wave * 32 + sub * 16 + hi * 8 + e;
                float v = acc[sub][j][e] * scale + bv;
                if (relu) v = fmaxf(v, 0.0f);
                if (Cf)  Cf[(size_t)m * Nt + n] = v;
                if (Cb)  Cb[(size_t)m * Nt + n] = (bf16_t)v;
                if (CbT) CbT[(size_t)n * M + m] = (bf16_t)v;  // transposed (V^T)
            }
        }
    }
}

// ---------------- masked row softmax: f32 scores + adj -> bf16 probs ----------------
__global__ void __launch_bounds__(256)
k_softmax(const float* __restrict__ S, const int* __restrict__ adj,
          bf16_t* __restrict__ P, int n) {
    const int row = blockIdx.x;
    const float* s = S + (size_t)row * n;
    const int*   a = adj + (size_t)row * n;
    bf16_t*      p = P + (size_t)row * n;

    __shared__ float red[256];
    const int tid = threadIdx.x;

    float mv[16];
    float m = -INFINITY;
    #pragma unroll
    for (int c = 0; c < 16; ++c) {
        int i = tid + c * 256;
        float v = (a[i] > 0) ? s[i] : -1.0e9f;   // matches reference mask fill
        mv[c] = v;
        m = fmaxf(m, v);
    }
    red[tid] = m; __syncthreads();
    for (int off = 128; off > 0; off >>= 1) {
        if (tid < off) red[tid] = fmaxf(red[tid], red[tid + off]);
        __syncthreads();
    }
    m = red[0]; __syncthreads();

    float sum = 0.0f;
    #pragma unroll
    for (int c = 0; c < 16; ++c) { mv[c] = __expf(mv[c] - m); sum += mv[c]; }
    red[tid] = sum; __syncthreads();
    for (int off = 128; off > 0; off >>= 1) {
        if (tid < off) red[tid] += red[tid + off];
        __syncthreads();
    }
    const float inv = 1.0f / red[0];
    #pragma unroll
    for (int c = 0; c < 16; ++c) p[tid + c * 256] = (bf16_t)(mv[c] * inv);
}

// ---------------------------------------------------------------------------

extern "C" void kernel_launch(void* const* d_in, const int* in_sizes, int n_in,
                              void* d_out, int out_size, void* d_ws, size_t ws_size,
                              hipStream_t stream) {
    const float* x     = (const float*)d_in[0];
    const int*   adj   = (const int*)d_in[1];
    const float* emb_w = (const float*)d_in[2];
    const float* emb_b = (const float*)d_in[3];
    const float* qw    = (const float*)d_in[4];
    const float* qb    = (const float*)d_in[5];
    const float* kw    = (const float*)d_in[6];
    const float* kb    = (const float*)d_in[7];
    const float* vw    = (const float*)d_in[8];
    const float* vb    = (const float*)d_in[9];
    const float* f1w   = (const float*)d_in[10];
    const float* f1b   = (const float*)d_in[11];
    const float* f2w   = (const float*)d_in[12];
    const float* f2b   = (const float*)d_in[13];
    const float* outw  = (const float*)d_in[14];
    const float* outb  = (const float*)d_in[15];
    float* out = (float*)d_out;

    // ---- workspace carve-out (256B aligned) ----
    size_t off = 0;
    char* base = (char*)d_ws;
    auto alloc = [&](size_t bytes) -> char* {
        off = (off + 255) & ~(size_t)255;
        char* p = base + off;
        off += bytes;
        return p;
    };
    bf16_t* x_bf   = (bf16_t*)alloc((size_t)N_NODES * DIN * 2);
    bf16_t* embwt  = (bf16_t*)alloc((size_t)DIN * DH * 2);
    bf16_t* qwt    = (bf16_t*)alloc((size_t)NLAYERS * DH * DH * 2);
    bf16_t* kwt    = (bf16_t*)alloc((size_t)NLAYERS * DH * DH * 2);
    bf16_t* vwt    = (bf16_t*)alloc((size_t)NLAYERS * DH * DH * 2);
    bf16_t* f1wt   = (bf16_t*)alloc((size_t)NLAYERS * DH * DH * 2);
    bf16_t* f2wt   = (bf16_t*)alloc((size_t)NLAYERS * DH * DH * 2);
    bf16_t* outwt  = (bf16_t*)alloc((size_t)DH * DOUT * 2);
    bf16_t* hA     = (bf16_t*)alloc((size_t)N_NODES * DH * 2);
    bf16_t* hB     = (bf16_t*)alloc((size_t)N_NODES * DH * 2);
    bf16_t* Qb     = (bf16_t*)alloc((size_t)N_NODES * DH * 2);
    bf16_t* Kb     = (bf16_t*)alloc((size_t)N_NODES * DH * 2);
    bf16_t* Vt     = (bf16_t*)alloc((size_t)DH * N_NODES * 2);  // V transposed [DH x N]
    bf16_t* Ob     = (bf16_t*)alloc((size_t)N_NODES * DH * 2);
    bf16_t* H1b    = (bf16_t*)alloc((size_t)N_NODES * DH * 2);
    float*  Sc     = (float*) alloc((size_t)N_NODES * N_NODES * 4);
    bf16_t* Pb     = (bf16_t*)alloc((size_t)N_NODES * N_NODES * 2);
    (void)ws_size; (void)in_sizes; (void)n_in; (void)out_size;

    const float scale = 1.0f / sqrtf((float)DH);

    // ---- stage weights/activations to bf16 (transposed for nt-GEMM) ----
    {
        int n = N_NODES * DIN;
        k_f32_to_bf16<<<(n + 255) / 256, 256, 0, stream>>>(x, x_bf, n);
        int pw = DH * DH;
        k_transpose_bf16<<<(DIN * DH + 255) / 256, 256, 0, stream>>>(emb_w, embwt, DIN, DH, 1);
        k_transpose_bf16<<<(NLAYERS * pw + 255) / 256, 256, 0, stream>>>(qw,  qwt,  DH, DH, NLAYERS);
        k_transpose_bf16<<<(NLAYERS * pw + 255) / 256, 256, 0, stream>>>(kw,  kwt,  DH, DH, NLAYERS);
        k_transpose_bf16<<<(NLAYERS * pw + 255) / 256, 256, 0, stream>>>(vw,  vwt,  DH, DH, NLAYERS);
        k_transpose_bf16<<<(NLAYERS * pw + 255) / 256, 256, 0, stream>>>(f1w, f1wt, DH, DH, NLAYERS);
        k_transpose_bf16<<<(NLAYERS * pw + 255) / 256, 256, 0, stream>>>(f2w, f2wt, DH, DH, NLAYERS);
        k_transpose_bf16<<<(DH * DOUT + 255) / 256, 256, 0, stream>>>(outw, outwt, DH, DOUT, 1);
    }

    dim3 blk(128);
    auto gemm = [&](const bf16_t* A, const bf16_t* Bt, const float* bias,
                    float* Cf, bf16_t* Cb, bf16_t* CbT,
                    int M, int Nt, int K, float sc, int relu, int addpe) {
        dim3 grd(M / 128, Nt / 64);
        k_gemm_nt<<<grd, blk, 0, stream>>>(A, Bt, bias, Cf, Cb, CbT, M, Nt, K, sc, relu, addpe);
    };

    // ---- embedding: h = x @ emb_w + emb_b + pe[0] ----
    gemm(x_bf, embwt, emb_b, nullptr, hA, nullptr, N_NODES, DH, DIN, 1.0f, 0, 1);

    bf16_t* hcur = hA;
    bf16_t* hnxt = hB;

    for (int l = 0; l < NLAYERS; ++l) {
        const size_t wo = (size_t)l * DH * DH;
        const size_t bo = (size_t)l * DH;
        // projections
        gemm(hcur, qwt + wo, qb + bo, nullptr, Qb, nullptr, N_NODES, DH, DH, 1.0f, 0, 0);
        gemm(hcur, kwt + wo, kb + bo, nullptr, Kb, nullptr, N_NODES, DH, DH, 1.0f, 0, 0);
        gemm(hcur, vwt + wo, vb + bo, nullptr, nullptr, Vt, N_NODES, DH, DH, 1.0f, 0, 0);
        // scores = (Q @ K^T) * scale   (f32)
        gemm(Qb, Kb, nullptr, Sc, nullptr, nullptr, N_NODES, N_NODES, DH, scale, 0, 0);
        // masked softmax -> bf16 probs
        k_softmax<<<N_NODES, 256, 0, stream>>>(Sc, adj, Pb, N_NODES);
        // O = P @ V  (V^T is the nt B operand)
        gemm(Pb, Vt, nullptr, nullptr, Ob, nullptr, N_NODES, DH, N_NODES, 1.0f, 0, 0);
        // FFN
        gemm(Ob,  f1wt + wo, f1b + bo, nullptr, H1b,  nullptr, N_NODES, DH, DH, 1.0f, 1, 0);
        gemm(H1b, f2wt + wo, f2b + bo, nullptr, hnxt, nullptr, N_NODES, DH, DH, 1.0f, 0, 0);
        bf16_t* t = hcur; hcur = hnxt; hnxt = t;
    }

    // ---- output projection (f32 straight to d_out) ----
    gemm(hcur, outwt, outb, out, nullptr, nullptr, N_NODES, DOUT, DH, 1.0f, 0, 0);
}
